// TreeLSTMParser_35897336660781
// MI455X (gfx1250) — matrix operand order repaired and checked by reading, not compiled, not run on hardware
//
#include <hip/hip_runtime.h>
#include <hip/hip_bf16.h>

// ---------------- problem constants ----------------
constexpr int N_NODES = 2048;
constexpr int D_IN    = 768;
constexpr int H       = 768;
constexpr int OUTC    = 128;
constexpr int NLVL    = 12;          // floor(log2(2048)) + 1
constexpr int G3      = 3 * H;       // 2304 (i,o,u)
constexpr int G4      = 4 * H;       // 3072 (i,f,g,o)
constexpr int CIN     = 2 * H;       // 1536
constexpr int FXROWS  = 2 * N_NODES; // padded rows for per-edge f-gate GEMM
constexpr int KMAX    = CIN;         // largest GEMM K (zero row length)
constexpr int SEQ_BLOCKS = 16;       // persistent blocks for recurrent LSTM
constexpr int HCHUNK  = H / SEQ_BLOCKS; // 48 hidden dims per block

// ---------------- types ----------------
typedef __attribute__((ext_vector_type(16))) __bf16    v16bf;
typedef __attribute__((ext_vector_type(8)))  float     v8f;
typedef __attribute__((ext_vector_type(4)))  unsigned  u32x4;

union FragU { v16bf v; u32x4 q[2]; };

// pack two f32 into one dword of bf16 (RNE)
__device__ __forceinline__ unsigned pk2bf(float lo, float hi) {
    union { float f; unsigned u; } a, b; a.f = lo; b.f = hi;
    unsigned ra = (a.u + 0x7FFFu + ((a.u >> 16) & 1u)) >> 16;
    unsigned rb = (b.u + 0x7FFFu + ((b.u >> 16) & 1u)) & 0xFFFF0000u;
    return (ra & 0xFFFFu) | rb;
}
__device__ __forceinline__ float bf2f(__bf16 b) {
    union { __bf16 b; unsigned short s; } y; y.b = b;
    union { unsigned u; float f; } x; x.u = ((unsigned)y.s) << 16;
    return x.f;
}
__device__ __forceinline__ float sigf(float x) { return 1.f / (1.f + __expf(-x)); }

// =====================================================================
// WMMA GEMM:  C[M x N](f32) = gather(A)[M x K](bf16) * W[N x K]^T(bf16) + bias
//  - invalid/padded A rows resolve to a zero dummy row => NO predicated loads
//  - fragments loaded as 2x b128 each; dual accumulators, K stepped by 64
// One 16x16 tile per wave, 8 waves / 256-thread block; grid = (M/16, N/128)
// =====================================================================
__global__ void gemm_wmma_bf16(const __bf16* __restrict__ A,
                               const int*    __restrict__ rowIdx,
                               int lda,
                               const __bf16* __restrict__ W,
                               const __bf16* __restrict__ zrow,   // [KMAX] zeros
                               const float*  __restrict__ bias,
                               const float*  __restrict__ bias2,
                               float* __restrict__ C, int ldc,
                               int M, int N, int K,
                               const int* __restrict__ mActPtr, int mActScale) {
    const int wave = threadIdx.x >> 5;
    const int lane = threadIdx.x & 31;
    const int tn   = blockIdx.y * 8 + wave;
    if (tn * 16 >= N) return;

    const int Mact = mActPtr ? (mActScale * (*mActPtr)) : M;
    const int m0   = blockIdx.x * 16;
    if (m0 >= Mact) return;                      // wave-uniform early exit

    const int half = lane >> 4;                  // 0: lanes 0-15, 1: lanes 16-31
    const int l16  = lane & 15;

    // Resolve this lane's A row; out-of-range / -1 rows -> zero dummy row.
    const __bf16* arow = zrow;
    {
        const int r = m0 + l16;
        if (r < Mact) {
            const int g = rowIdx ? rowIdx[r] : r;
            if (g >= 0) arow = A + (size_t)g * lda;
        }
    }
    const int ncol = tn * 16 + l16;
    const __bf16* wrow = W + (size_t)ncol * K;

    v8f acc0 = {}, acc1 = {};
    const int abase = half * 8;                  // A: lanes hi-half hold K+8
    const int bbase = half * 8;                  // B: lanes hi-half hold K+8
    for (int kk = 0; kk < K; kk += 64) {
        FragU a0, b0, a1, b1;
        // A 16x32 bf16 fragment: elems 0-7 -> K=abase.., elems 8-15 -> K=abase+16..
        a0.q[0] = *(const u32x4*)(arow + kk + abase);
        a0.q[1] = *(const u32x4*)(arow + kk + abase + 16);
        a1.q[0] = *(const u32x4*)(arow + kk + 32 + abase);
        a1.q[1] = *(const u32x4*)(arow + kk + 32 + abase + 16);
        // B 32x16 bf16 fragment: elems 0-7 -> K=bbase.., elems 8-15 -> K=bbase+16..
        b0.q[0] = *(const u32x4*)(wrow + kk + bbase);
        b0.q[1] = *(const u32x4*)(wrow + kk + bbase + 16);
        b1.q[0] = *(const u32x4*)(wrow + kk + 32 + bbase);
        b1.q[1] = *(const u32x4*)(wrow + kk + 32 + bbase + 16);

        acc0 = __builtin_amdgcn_wmma_f32_16x16x32_bf16(
            false, a0.v, false, b0.v, (short)0, acc0, false, false);
        acc1 = __builtin_amdgcn_wmma_f32_16x16x32_bf16(
            false, a1.v, false, b1.v, (short)0, acc1, false, false);
    }

    const float bb = (bias ? bias[ncol] : 0.f) + (bias2 ? bias2[ncol] : 0.f);
#pragma unroll
    for (int r = 0; r < 8; ++r) {
        const int row = m0 + half * 8 + r;       // C: VGPR r -> M=r (+8 hi half)
        C[(size_t)row * ldc + ncol] = acc0[r] + acc1[r] + bb;
    }
}

// =====================================================================
// Utility kernels
// =====================================================================
__global__ void zero_f32(float* p, int n) {
    int i = blockIdx.x * blockDim.x + threadIdx.x;
    if (i < n) p[i] = 0.f;
}
__global__ void zero_i32(int* p, int n) {
    int i = blockIdx.x * blockDim.x + threadIdx.x;
    if (i < n) p[i] = 0;
}
// f32 -> packed bf16 (n must be even; npairs = n/2)
__global__ void cvt_f32_bf16(const float* __restrict__ s, __bf16* __restrict__ d,
                             int npairs) {
    int i = blockIdx.x * blockDim.x + threadIdx.x;
    if (i < npairs) {
        float2 p = ((const float2*)s)[i];
        ((unsigned*)d)[i] = pk2bf(p.x, p.y);
    }
}

// =====================================================================
// Build per-level node buckets + per-node child lists (1 block).
// =====================================================================
__global__ void build_tree_meta(const int* __restrict__ node_order,
                                const int* __restrict__ adj,        // (E,2) int32
                                int* __restrict__ lvl_cnt, int* __restrict__ lvl_off,
                                int* __restrict__ lvl_nodes,
                                int* __restrict__ child_cnt, int* __restrict__ child_idx,
                                int* __restrict__ cursors) {
    const int tid = threadIdx.x;
    for (int i = tid; i < NLVL; i += blockDim.x) { lvl_cnt[i] = 0; cursors[i] = 0; }
    for (int i = tid; i < N_NODES; i += blockDim.x) child_cnt[i] = 0;
    __syncthreads();
    for (int i = tid; i < N_NODES; i += blockDim.x)
        atomicAdd(&lvl_cnt[node_order[i]], 1);
    for (int e = tid; e < N_NODES - 1; e += blockDim.x) {
        const int p = adj[2 * e], c = adj[2 * e + 1];
        const int s = atomicAdd(&child_cnt[p], 1);
        child_idx[2 * p + s] = c;
    }
    __syncthreads();
    if (tid == 0) {
        int acc = 0;
        for (int l = 0; l < NLVL; ++l) { lvl_off[l] = acc; acc += lvl_cnt[l]; }
        lvl_off[NLVL] = acc;
    }
    __syncthreads();
    for (int i = tid; i < N_NODES; i += blockDim.x) {
        const int l = node_order[i];
        lvl_nodes[lvl_off[l] + atomicAdd(&cursors[l], 1)] = i;
    }
}

// =====================================================================
// Per-level gather: h_sum(bf16) over children + child-row gather index.
// grid = N_NODES blocks (early exit), block = 256
// =====================================================================
__global__ void tree_gather(int lvl,
                            const int* __restrict__ lvl_cnt,
                            const int* __restrict__ lvl_off,
                            const int* __restrict__ lvl_nodes,
                            const int* __restrict__ child_cnt,
                            const int* __restrict__ child_idx,
                            const __bf16* __restrict__ hb,      // [N][H] bf16
                            __bf16* __restrict__ h_sum,         // [N][H] bf16 compact
                            int* __restrict__ frow) {           // [2N]
    const int s = blockIdx.x;
    if (s >= lvl_cnt[lvl]) return;
    const int p  = lvl_nodes[lvl_off[lvl] + s];
    const int nc = child_cnt[p];
    const int c0 = (nc > 0) ? child_idx[2 * p]     : -1;
    const int c1 = (nc > 1) ? child_idx[2 * p + 1] : -1;
    if (threadIdx.x == 0) { frow[2 * s] = c0; frow[2 * s + 1] = c1; }
    unsigned* dst = (unsigned*)(h_sum + (size_t)s * H);
    const __bf16* r0 = (c0 >= 0) ? hb + (size_t)c0 * H : nullptr;
    const __bf16* r1 = (c1 >= 0) ? hb + (size_t)c1 * H : nullptr;
    for (int j = threadIdx.x; j < H / 2; j += blockDim.x) {
        float v0 = 0.f, v1 = 0.f;
        if (r0) { v0 += bf2f(r0[2 * j]); v1 += bf2f(r0[2 * j + 1]); }
        if (r1) { v0 += bf2f(r1[2 * j]); v1 += bf2f(r1[2 * j + 1]); }
        dst[j] = pk2bf(v0, v1);
    }
}

// =====================================================================
// Per-level pointwise update: gates, f-weighted child c sum; writes c(f32),
// h(bf16). Parents at level lvl are disjoint from their children (no race).
// =====================================================================
__global__ void tree_update(int lvl,
                            const int* __restrict__ lvl_cnt,
                            const int* __restrict__ lvl_off,
                            const int* __restrict__ lvl_nodes,
                            const int* __restrict__ child_cnt,
                            const int* __restrict__ child_idx,
                            const float* __restrict__ x_iou,   // [N][3H]
                            const float* __restrict__ x_f,     // [N][H]
                            const float* __restrict__ iou_act, // [N][3H] compact
                            const float* __restrict__ fx,      // [2N][H]
                            __bf16* __restrict__ hb,           // [N][H] bf16
                            float*  __restrict__ c) {          // [N][H] f32
    const int s = blockIdx.x;
    if (s >= lvl_cnt[lvl]) return;
    const int p  = lvl_nodes[lvl_off[lvl] + s];
    const int nc = child_cnt[p];
    const int c0 = (nc > 0) ? child_idx[2 * p]     : -1;
    const int c1 = (nc > 1) ? child_idx[2 * p + 1] : -1;
    const float2* xi = (const float2*)(x_iou + (size_t)p * G3);
    const float2* ia = (const float2*)(iou_act + (size_t)s * G3);
    const float2* xf = (const float2*)(x_f + (size_t)p * H);
    const float2* f0 = (const float2*)(fx + (size_t)(2 * s) * H);
    const float2* f1 = (const float2*)(fx + (size_t)(2 * s + 1) * H);
    const float2* cc0 = (c0 >= 0) ? (const float2*)(c + (size_t)c0 * H) : nullptr;
    const float2* cc1 = (c1 >= 0) ? (const float2*)(c + (size_t)c1 * H) : nullptr;
    float2*   cp = (float2*)(c + (size_t)p * H);
    unsigned* hp = (unsigned*)(hb + (size_t)p * H);
    const int HP = H / 2;
    for (int j = threadIdx.x; j < HP; j += blockDim.x) {
        const float2 vi = xi[j],          ai = ia[j];
        const float2 vo = xi[HP + j],     ao = ia[HP + j];
        const float2 vu = xi[2 * HP + j], au = ia[2 * HP + j];
        const float i0 = sigf(vi.x + ai.x),  i1 = sigf(vi.y + ai.y);
        const float o0 = sigf(vo.x + ao.x),  o1 = sigf(vo.y + ao.y);
        const float u0 = tanhf(vu.x + au.x), u1 = tanhf(vu.y + au.y);
        const float2 vf = xf[j];
        float cs0 = 0.f, cs1 = 0.f;
        if (cc0) {
            const float2 fg = f0[j], cv = cc0[j];
            cs0 += sigf(vf.x + fg.x) * cv.x; cs1 += sigf(vf.y + fg.y) * cv.y;
        }
        if (cc1) {
            const float2 fg = f1[j], cv = cc1[j];
            cs0 += sigf(vf.x + fg.x) * cv.x; cs1 += sigf(vf.y + fg.y) * cv.y;
        }
        const float cn0 = i0 * u0 + cs0, cn1 = i1 * u1 + cs1;
        float2 co; co.x = cn0; co.y = cn1;
        cp[j] = co;
        hp[j] = pk2bf(o0 * tanhf(cn0), o1 * tanhf(cn1));
    }
}

// =====================================================================
// Persistent recurrent LSTM: 16 blocks, device-wide generation barrier
// per step. Block b owns hidden dims [48b, 48b+48). hs[t] doubles as the
// h-state exchange buffer. xg already holds b_ih + b_hh.
// =====================================================================
__global__ void seq_lstm(const float* __restrict__ xg,    // [N][4H]
                         const float* __restrict__ W_hh,  // [4H][H]
                         float* __restrict__ hs,          // [N][H]
                         int* __restrict__ bar) {
    __shared__ float hsh[H];
    __shared__ float gl[4 * HCHUNK];
    __shared__ float csh[HCHUNK];
    const int b = blockIdx.x, tid = threadIdx.x;
    const int d0 = b * HCHUNK;
    if (tid < HCHUNK) csh[tid] = 0.f;
    __syncthreads();

    for (int t = 0; t < N_NODES; ++t) {
        if (t == 0) { for (int k = tid; k < H; k += blockDim.x) hsh[k] = 0.f; }
        else        { for (int k = tid; k < H; k += blockDim.x) hsh[k] = hs[(size_t)(t - 1) * H + k]; }
        __syncthreads();

        if (tid == 0 && t + 1 < N_NODES)
            __builtin_prefetch(xg + (size_t)(t + 1) * G4, 0, 1);

        if (tid < 4 * HCHUNK) {                  // 192 gate dots of length 768
            const int gate = tid / HCHUNK;
            const int row  = gate * H + d0 + (tid % HCHUNK);
            const float4* wr = (const float4*)(W_hh + (size_t)row * H); // L2-resident
            float acc = xg[(size_t)t * G4 + row];
            for (int k = 0; k < H / 4; ++k) {
                const float4 w = wr[k];
                acc = fmaf(w.x, hsh[4 * k],     acc);
                acc = fmaf(w.y, hsh[4 * k + 1], acc);
                acc = fmaf(w.z, hsh[4 * k + 2], acc);
                acc = fmaf(w.w, hsh[4 * k + 3], acc);
            }
            gl[tid] = acc;
        }
        __syncthreads();

        if (tid < HCHUNK) {                      // i, f, g, o
            const float iv = sigf (gl[tid]);
            const float fv = sigf (gl[HCHUNK + tid]);
            const float gv = tanhf(gl[2 * HCHUNK + tid]);
            const float ov = sigf (gl[3 * HCHUNK + tid]);
            const float cn = fv * csh[tid] + iv * gv;
            csh[tid] = cn;
            hs[(size_t)t * H + d0 + tid] = ov * tanhf(cn);
        }
        __syncthreads();
        if (tid == 0) {                          // device-wide step barrier
            __threadfence();
            atomicAdd(bar, 1);
            const int target = SEQ_BLOCKS * (t + 1);
            while (atomicAdd(bar, 0) < target) {}
            __threadfence();
        }
        __syncthreads();
    }
}

// =====================================================================
// Concat [tree_h(bf16) | lstm_h(f32->bf16)] -> cls_bf; grid = N_NODES
// =====================================================================
__global__ void concat_rows(const __bf16* __restrict__ th,
                            const float*  __restrict__ sh,
                            __bf16* __restrict__ cls) {
    const int i = blockIdx.x;
    unsigned* dst = (unsigned*)(cls + (size_t)i * CIN);
    const unsigned* src = (const unsigned*)(th + (size_t)i * H);
    const float2*   sf  = (const float2*)(sh + (size_t)i * H);
    for (int j = threadIdx.x; j < H / 2; j += blockDim.x) {
        dst[j] = src[j];
        const float2 p = sf[j];
        dst[H / 2 + j] = pk2bf(p.x, p.y);
    }
}

// =====================================================================
// log_softmax over axis=0 (columns): one block per output class.
// =====================================================================
__global__ void logsoftmax_col(const float* __restrict__ logits, float* __restrict__ out) {
    __shared__ float red[256];
    const int j = blockIdx.x, tid = threadIdx.x;
    float m = -3.4e38f;
    for (int i = tid; i < N_NODES; i += 256) m = fmaxf(m, logits[(size_t)i * OUTC + j]);
    red[tid] = m; __syncthreads();
    for (int s = 128; s > 0; s >>= 1) { if (tid < s) red[tid] = fmaxf(red[tid], red[tid + s]); __syncthreads(); }
    const float mx = red[0]; __syncthreads();
    float ssum = 0.f;
    for (int i = tid; i < N_NODES; i += 256) ssum += __expf(logits[(size_t)i * OUTC + j] - mx);
    red[tid] = ssum; __syncthreads();
    for (int s = 128; s > 0; s >>= 1) { if (tid < s) red[tid] += red[tid + s]; __syncthreads(); }
    const float lse = mx + __logf(red[0]);
    for (int i = tid; i < N_NODES; i += 256) out[(size_t)i * OUTC + j] = logits[(size_t)i * OUTC + j] - lse;
}

// =====================================================================
// Host side
// =====================================================================
extern "C" void kernel_launch(void* const* d_in, const int* in_sizes, int n_in,
                              void* d_out, int out_size, void* d_ws, size_t ws_size,
                              hipStream_t stream) {
    const float* tree_features   = (const float*)d_in[0];
    const float* features_tensor = (const float*)d_in[1];
    const float* W_iou = (const float*)d_in[2];
    const float* b_iou = (const float*)d_in[3];
    const float* U_iou = (const float*)d_in[4];
    const float* W_f   = (const float*)d_in[5];
    const float* b_f   = (const float*)d_in[6];
    const float* U_f   = (const float*)d_in[7];
    const float* W_ih  = (const float*)d_in[8];
    const float* b_ih  = (const float*)d_in[9];
    const float* W_hh  = (const float*)d_in[10];
    const float* b_hh  = (const float*)d_in[11];
    const float* W_cls = (const float*)d_in[12];
    const float* b_cls = (const float*)d_in[13];
    const int* node_order = (const int*)d_in[14];
    const int* adjacency  = (const int*)d_in[15];

    // ---- workspace layout: f32 region, bf16 region, int region ----
    float* ws = (float*)d_ws;
    size_t off = 0;
    auto falloc = [&](size_t n) { float* p = ws + off; off += n; return p; };
    float* x_iou   = falloc((size_t)N_NODES * G3);
    float* x_f     = falloc((size_t)N_NODES * H);
    float* x_ih    = falloc((size_t)N_NODES * G4);
    float* c       = falloc((size_t)N_NODES * H);
    float* iou_act = falloc((size_t)N_NODES * G3);
    float* fx      = falloc((size_t)FXROWS * H);
    float* hs      = falloc((size_t)N_NODES * H);
    float* logits  = falloc((size_t)N_NODES * OUTC);

    __bf16* bbase = (__bf16*)(ws + off);
    size_t boff = 0;
    auto balloc = [&](size_t n) { __bf16* p = bbase + boff; boff += n; return p; };
    __bf16* a_tree  = balloc((size_t)N_NODES * D_IN);
    __bf16* a_feat  = balloc((size_t)N_NODES * D_IN);
    __bf16* wb_iou  = balloc((size_t)G3 * D_IN);
    __bf16* wb_f    = balloc((size_t)H * D_IN);
    __bf16* wb_ih   = balloc((size_t)G4 * D_IN);
    __bf16* ub_iou  = balloc((size_t)G3 * H);
    __bf16* ub_f    = balloc((size_t)H * H);
    __bf16* wb_cls  = balloc((size_t)OUTC * CIN);
    __bf16* h_bf    = balloc((size_t)N_NODES * H);
    __bf16* hsum_bf = balloc((size_t)N_NODES * H);
    __bf16* cls_bf  = balloc((size_t)N_NODES * CIN);
    __bf16* zrow    = balloc((size_t)KMAX);

    int* ip = (int*)(bbase + boff);
    int* lvl_cnt   = ip;  ip += NLVL;
    int* lvl_off   = ip;  ip += NLVL + 1;
    int* cursors   = ip;  ip += NLVL;
    int* lvl_nodes = ip;  ip += N_NODES;
    int* child_cnt = ip;  ip += N_NODES;
    int* child_idx = ip;  ip += 2 * N_NODES;
    int* frow      = ip;  ip += FXROWS;
    int* bar       = ip;  ip += 1;

    // ---- init: c=0, h_bf=0, zrow=0, barrier=0 ----
    zero_f32<<<(N_NODES * H + 255) / 256, 256, 0, stream>>>(c, N_NODES * H);
    zero_i32<<<(N_NODES * H / 2 + 255) / 256, 256, 0, stream>>>((int*)h_bf, N_NODES * H / 2);
    zero_i32<<<(KMAX / 2 + 31) / 32, 32, 0, stream>>>((int*)zrow, KMAX / 2);
    zero_i32<<<1, 32, 0, stream>>>(bar, 1);

    build_tree_meta<<<1, 256, 0, stream>>>(node_order, adjacency, lvl_cnt, lvl_off,
                                           lvl_nodes, child_cnt, child_idx, cursors);

    // ---- one-time f32 -> bf16 conversions (weights + dense activations) ----
    auto cvt = [&](const float* s, __bf16* d, size_t n) {
        const int np = (int)(n / 2);
        cvt_f32_bf16<<<(np + 255) / 256, 256, 0, stream>>>(s, d, np);
    };
    cvt(tree_features,   a_tree, (size_t)N_NODES * D_IN);
    cvt(features_tensor, a_feat, (size_t)N_NODES * D_IN);
    cvt(W_iou, wb_iou, (size_t)G3 * D_IN);
    cvt(W_f,   wb_f,   (size_t)H * D_IN);
    cvt(W_ih,  wb_ih,  (size_t)G4 * D_IN);
    cvt(U_iou, ub_iou, (size_t)G3 * H);
    cvt(U_f,   ub_f,   (size_t)H * H);
    cvt(W_cls, wb_cls, (size_t)OUTC * CIN);

    // ---- input projections (bf16 WMMA, f32 accumulate) ----
    gemm_wmma_bf16<<<dim3(N_NODES / 16, G3 / 128), 256, 0, stream>>>(
        a_tree, nullptr, D_IN, wb_iou, zrow, b_iou, nullptr,
        x_iou, G3, N_NODES, G3, D_IN, nullptr, 1);
    gemm_wmma_bf16<<<dim3(N_NODES / 16, H / 128), 256, 0, stream>>>(
        a_tree, nullptr, D_IN, wb_f, zrow, b_f, nullptr,
        x_f, H, N_NODES, H, D_IN, nullptr, 1);
    gemm_wmma_bf16<<<dim3(N_NODES / 16, G4 / 128), 256, 0, stream>>>(
        a_feat, nullptr, D_IN, wb_ih, zrow, b_ih, b_hh,
        x_ih, G4, N_NODES, G4, D_IN, nullptr, 1);

    // ---- TreeLSTM: 12 levels, active nodes only ----
    for (int lvl = 0; lvl < NLVL; ++lvl) {
        tree_gather<<<N_NODES, 256, 0, stream>>>(lvl, lvl_cnt, lvl_off, lvl_nodes,
                                                 child_cnt, child_idx, h_bf, hsum_bf, frow);
        gemm_wmma_bf16<<<dim3(N_NODES / 16, G3 / 128), 256, 0, stream>>>(
            hsum_bf, nullptr, H, ub_iou, zrow, nullptr, nullptr,
            iou_act, G3, N_NODES, G3, H, lvl_cnt + lvl, 1);
        gemm_wmma_bf16<<<dim3(FXROWS / 16, H / 128), 256, 0, stream>>>(
            h_bf, frow, H, ub_f, zrow, nullptr, nullptr,
            fx, H, FXROWS, H, H, lvl_cnt + lvl, 2);
        tree_update<<<N_NODES, 256, 0, stream>>>(lvl, lvl_cnt, lvl_off, lvl_nodes,
                                                 child_cnt, child_idx,
                                                 x_iou, x_f, iou_act, fx, h_bf, c);
    }

    // ---- sequential LSTM (persistent, device-wide step barrier) ----
    seq_lstm<<<SEQ_BLOCKS, 256, 0, stream>>>(x_ih, W_hh, hs, bar);

    // ---- classifier + log_softmax over axis 0 ----
    concat_rows<<<N_NODES, 256, 0, stream>>>(h_bf, hs, cls_bf);
    gemm_wmma_bf16<<<dim3(N_NODES / 16, OUTC / 128), 256, 0, stream>>>(
        cls_bf, nullptr, CIN, wb_cls, zrow, b_cls, nullptr,
        logits, OUTC, N_NODES, OUTC, CIN, nullptr, 1);
    logsoftmax_col<<<OUTC, 256, 0, stream>>>(logits, (float*)d_out);

    (void)in_sizes; (void)n_in; (void)out_size; (void)ws_size;
}
